// SE_GATWithEdgeAttr_23244363006449
// MI455X (gfx1250) — compile-verified, matrix operand order checked by reading, not run on hardware
//
#include <hip/hip_runtime.h>
#include <hip/hip_bf16.h>
#include <math.h>

#define N_NODES    100000
#define N_EDGES    1600000
#define NUM_FEAT   32
#define EDGE_DIM   8
#define GAT_DIM    16
#define HEADS      4
#define HID        64
#define OUT_DIM    2
#define NUM_GRAPHS 64
#define NEG_SLOPE  0.2f

typedef _Float16 half8_t  __attribute__((ext_vector_type(8)));
typedef _Float16 half16_t __attribute__((ext_vector_type(16)));
typedef float    f32x8_t  __attribute__((ext_vector_type(8)));

// ---------------------------------------------------------------- helpers
__device__ __forceinline__ void atomicMaxF32(float* addr, float val) {
    // sign-aware int trick; works with -inf init for mixed signs
    if (val >= 0.0f) atomicMax((int*)addr, __float_as_int(val));
    else             atomicMin((unsigned int*)addr, __float_as_uint(val));
}

// ---------------------------------------------------------------- prep
__global__ void k_edge_degree(const int* __restrict__ dst,
                              const float* __restrict__ ea,
                              float* __restrict__ cnt, float* __restrict__ sumA, int E) {
    int e = blockIdx.x * blockDim.x + threadIdx.x;
    if (e >= E) return;
    int d = dst[e];
    atomicAdd(&cnt[d], 1.0f);
    const float* a = ea + (size_t)e * EDGE_DIM;
    float* s = sumA + (size_t)d * EDGE_DIM;
#pragma unroll
    for (int k = 0; k < EDGE_DIM; ++k) atomicAdd(&s[k], a[k]);
}

__global__ void k_mean_attr(const float* __restrict__ sumA, const float* __restrict__ cnt,
                            float* __restrict__ meanA, int n) {
    int i = blockIdx.x * blockDim.x + threadIdx.x;
    if (i >= n * EDGE_DIM) return;
    float c = fmaxf(cnt[i >> 3], 1.0f);
    meanA[i] = sumA[i] / c;
}

__global__ void k_f32_to_f16(const float* __restrict__ in, _Float16* __restrict__ out, int n) {
    int i = blockIdx.x * blockDim.x + threadIdx.x;
    if (i < n) out[i] = (_Float16)in[i];
}

// Wt[n][k] = (f16) W[k][n]     (W is [Kdim][64] row-major)
__global__ void k_prep_wt(const float* __restrict__ W, _Float16* __restrict__ Wt, int Kdim) {
    int i = blockIdx.x * blockDim.x + threadIdx.x;
    if (i >= 64 * Kdim) return;
    int n = i / Kdim, k = i % Kdim;
    Wt[n * Kdim + k] = (_Float16)W[k * 64 + n];
}

// w_eff[k][h] = sum_c We[k][h*16+c] * a_e[h][c]   (folds edge-proj + att_e dot)
__global__ void k_prep_weff(const float* __restrict__ We, const float* __restrict__ ae,
                            float* __restrict__ weff) {
    int t = threadIdx.x;
    if (t >= EDGE_DIM * HEADS) return;
    int k = t >> 2, h = t & 3;
    float s = 0.0f;
#pragma unroll
    for (int c = 0; c < GAT_DIM; ++c) s += We[k * HID + h * GAT_DIM + c] * ae[h * GAT_DIM + c];
    weff[k * HEADS + h] = s;
}

// ---------------------------------------------------------------- WMMA GEMM
// Y[N][64] = Ah[N][Kdim](f16) @ W[Kdim][64] with Wt(f16, col-major of W) in LDS.
// One wave computes a 16-row tile x 64 cols (4 accumulators), K-loop step 32.
__global__ void k_gemm_wmma(const _Float16* __restrict__ Ah,
                            const _Float16* __restrict__ WtG,
                            float* __restrict__ Y, int nRows, int Kdim) {
    __shared__ _Float16 sWt[64 * 64];
    int t = threadIdx.x;
    for (int i = t; i < 64 * Kdim; i += blockDim.x) sWt[i] = WtG[i];
    __syncthreads();

    int wave = t >> 5;
    int lane = t & 31;
    int tile = blockIdx.x * 8 + wave;           // 16-row tile
    if (tile * 16 >= nRows) return;
    int row0 = tile * 16;
    int m16  = lane & 15;
    int hi   = lane >> 4;

    f32x8_t acc[4] = {};
    for (int k0 = 0; k0 < Kdim; k0 += 32) {
        // A fragment (16-bit 16x32 layout): two contiguous 16B chunks per lane
        const _Float16* ap = Ah + (size_t)(row0 + m16) * Kdim + k0 + hi * 8;
        half8_t alo = *(const half8_t*)(ap);
        half8_t ahi = *(const half8_t*)(ap + 16);
        half16_t afrag = __builtin_shufflevector(alo, ahi,
            0,1,2,3,4,5,6,7,8,9,10,11,12,13,14,15);
#pragma unroll
        for (int nt = 0; nt < 4; ++nt) {
            const _Float16* bp = sWt + (nt * 16 + m16) * Kdim + k0 + hi * 8;
            half8_t blo = *(const half8_t*)(bp);
            half8_t bhi = *(const half8_t*)(bp + 16);
            half16_t bfrag = __builtin_shufflevector(blo, bhi,
                0,1,2,3,4,5,6,7,8,9,10,11,12,13,14,15);
            acc[nt] = __builtin_amdgcn_wmma_f32_16x16x32_f16(
                false, afrag, false, bfrag, (short)0, acc[nt], false, false);
        }
    }
    // C layout: lane<16 -> M=r, lane>=16 -> M=8+r ; N = lane&15
#pragma unroll
    for (int nt = 0; nt < 4; ++nt)
#pragma unroll
        for (int r = 0; r < 8; ++r) {
            int m = r + 8 * hi;
            Y[(size_t)(row0 + m) * HID + nt * 16 + m16] = acc[nt][r];
        }
}

// ---------------------------------------------------------------- attention
__global__ void k_node_logits(const float* __restrict__ xp,
                              const float* __restrict__ as, const float* __restrict__ ad,
                              float* __restrict__ als, float* __restrict__ ald, int n) {
    int i = blockIdx.x * blockDim.x + threadIdx.x;
    if (i >= n) return;
    const float* xr = xp + (size_t)i * HID;
#pragma unroll
    for (int h = 0; h < HEADS; ++h) {
        float s = 0.0f, d = 0.0f;
#pragma unroll
        for (int c = 0; c < GAT_DIM; ++c) {
            float v = xr[h * GAT_DIM + c];
            s += v * as[h * GAT_DIM + c];
            d += v * ad[h * GAT_DIM + c];
        }
        als[i * HEADS + h] = s;
        ald[i * HEADS + h] = d;
    }
}

__global__ void k_layer_init(float* __restrict__ m, float* __restrict__ den,
                             float* __restrict__ agg, int n) {
    int i = blockIdx.x * blockDim.x + threadIdx.x;
    if (i >= n * HID) return;
    agg[i] = 0.0f;
    if (i < n * HEADS) { m[i] = -INFINITY; den[i] = 0.0f; }
}

// pass A: alpha = leaky_relu(al_s[src]+al_d[dst]+al_e), segment max over dst
__global__ void k_edge_alpha(const int* __restrict__ src, const int* __restrict__ dst,
                             const float* __restrict__ ea, const float* __restrict__ meanA,
                             const float* __restrict__ als, const float* __restrict__ ald,
                             const float* __restrict__ weff,
                             float* __restrict__ alpha, float* __restrict__ m,
                             int E, int Etot) {
    int e = blockIdx.x * blockDim.x + threadIdx.x;
    if (e >= Etot) return;
    int s, d; const float* a;
    if (e < E) { s = src[e]; d = dst[e]; a = ea + (size_t)e * EDGE_DIM; }
    else       { s = d = e - E;          a = meanA + (size_t)(e - E) * EDGE_DIM; }
    float av[EDGE_DIM];
#pragma unroll
    for (int k = 0; k < EDGE_DIM; ++k) av[k] = a[k];
#pragma unroll
    for (int h = 0; h < HEADS; ++h) {
        float ale = 0.0f;
#pragma unroll
        for (int k = 0; k < EDGE_DIM; ++k) ale += av[k] * weff[k * HEADS + h];
        float v = als[s * HEADS + h] + ald[d * HEADS + h] + ale;
        v = v > 0.0f ? v : NEG_SLOPE * v;
        alpha[(size_t)e * HEADS + h] = v;
        atomicMaxF32(&m[d * HEADS + h], v);
    }
}

// pass B: ex = exp(alpha - m[dst]); den[dst] += ex   (in-place on alpha buffer)
__global__ void k_edge_exp(const int* __restrict__ dst, float* __restrict__ alpha,
                           const float* __restrict__ m, float* __restrict__ den,
                           int E, int Etot) {
    int e = blockIdx.x * blockDim.x + threadIdx.x;
    if (e >= Etot) return;
    int d = (e < E) ? dst[e] : e - E;
#pragma unroll
    for (int h = 0; h < HEADS; ++h) {
        float ex = __expf(alpha[(size_t)e * HEADS + h] - m[d * HEADS + h]);
        alpha[(size_t)e * HEADS + h] = ex;
        atomicAdd(&den[d * HEADS + h], ex);
    }
}

// pass C: out[dst] += xp[src] * (ex/(den[dst]+eps)); 64 channel-threads per edge
__global__ void k_edge_aggregate(const int* __restrict__ src, const int* __restrict__ dst,
                                 const float* __restrict__ ex, const float* __restrict__ den,
                                 const float* __restrict__ xp, float* __restrict__ agg,
                                 int E, int Etot) {
    long long tid = (long long)blockIdx.x * blockDim.x + threadIdx.x;
    int e = (int)(tid >> 6);
    if (e >= Etot) return;
    int c = (int)(tid & 63);
    int h = c >> 4;
    int s = (e < E) ? src[e] : e - E;
    int d = (e < E) ? dst[e] : e - E;
    float coef = ex[(size_t)e * HEADS + h] / (den[d * HEADS + h] + 1e-16f);
    atomicAdd(&agg[(size_t)d * HID + c], xp[(size_t)s * HID + c] * coef);
}

__global__ void k_finalize(const float* __restrict__ agg, const float* __restrict__ b,
                           float* __restrict__ h32, _Float16* __restrict__ h16,
                           int n, int doRelu) {
    int i = blockIdx.x * blockDim.x + threadIdx.x;
    if (i >= n * HID) return;
    float v = agg[i] + b[i & 63];
    if (doRelu) v = fmaxf(v, 0.0f);
    h32[i] = v;
    h16[i] = (_Float16)v;
}

// ---------------------------------------------------------------- pooling + MLP
__global__ void k_pool(const float* __restrict__ h32, const int* __restrict__ batch,
                       float* __restrict__ pooled, float* __restrict__ gcnt, int n) {
    int i = blockIdx.x * blockDim.x + threadIdx.x;
    if (i >= n * HID) return;
    int node = i >> 6, c = i & 63;
    int g = batch[node];
    atomicAdd(&pooled[g * HID + c], h32[i]);
    if (c == 0) atomicAdd(&gcnt[g], 1.0f);
}

__global__ void k_mlp_head(const float* __restrict__ pooled, const float* __restrict__ gcnt,
                           const float* __restrict__ fc1w, const float* __restrict__ fc1b,
                           const float* __restrict__ fc2w, const float* __restrict__ fc2b,
                           float* __restrict__ y) {
    int g = threadIdx.x;
    if (g >= NUM_GRAPHS) return;
    float p[HID];
    float c = fmaxf(gcnt[g], 1.0f);
#pragma unroll
    for (int i = 0; i < HID; ++i) p[i] = pooled[g * HID + i] / c;
    float z[2 * GAT_DIM];
    for (int j = 0; j < 2 * GAT_DIM; ++j) {
        float s = fc1b[j];
        for (int i = 0; i < HID; ++i) s += p[i] * fc1w[i * (2 * GAT_DIM) + j];
        z[j] = fmaxf(s, 0.0f);
    }
#pragma unroll
    for (int o = 0; o < OUT_DIM; ++o) {
        float s = fc2b[o];
        for (int j = 0; j < 2 * GAT_DIM; ++j) s += z[j] * fc2w[j * OUT_DIM + o];
        y[g * OUT_DIM + o] = s;
    }
}

// ---------------------------------------------------------------- launch
extern "C" void kernel_launch(void* const* d_in, const int* in_sizes, int n_in,
                              void* d_out, int out_size, void* d_ws, size_t ws_size,
                              hipStream_t stream) {
    const float* x        = (const float*)d_in[0];
    const int*   ei       = (const int*)d_in[1];
    const float* ea       = (const float*)d_in[2];
    const int*   batch    = (const int*)d_in[3];
    const float* W0       = (const float*)d_in[4];
    const float* as0      = (const float*)d_in[5];
    const float* ad0      = (const float*)d_in[6];
    const float* We0      = (const float*)d_in[7];
    const float* ae0      = (const float*)d_in[8];
    const float* b0       = (const float*)d_in[9];
    const float* Wh       = (const float*)d_in[10];
    const float* as_h     = (const float*)d_in[11];
    const float* ad_h     = (const float*)d_in[12];
    const float* We_h     = (const float*)d_in[13];
    const float* ae_h     = (const float*)d_in[14];
    const float* bh       = (const float*)d_in[15];
    const float* fc1w     = (const float*)d_in[16];
    const float* fc1b     = (const float*)d_in[17];
    const float* fc2w     = (const float*)d_in[18];
    const float* fc2b     = (const float*)d_in[19];
    float* y_out          = (float*)d_out;

    const int N = in_sizes[0] / NUM_FEAT;          // 100000
    const int E = in_sizes[1] / 2;                 // 1600000
    const int Etot = E + N;
    const int* src = ei;
    const int* dst = ei + E;

    // ---- workspace layout
    size_t off = 0;
    auto alloc = [&](size_t bytes) -> char* {
        char* p = (char*)d_ws + off;
        off += (bytes + 255) & ~(size_t)255;
        return p;
    };
    float*    hF32   = (float*)   alloc((size_t)N * HID * 4);
    _Float16* hF16   = (_Float16*)alloc((size_t)N * HID * 2);
    _Float16* xh16   = (_Float16*)alloc((size_t)N * NUM_FEAT * 2);
    float*    xp     = (float*)   alloc((size_t)N * HID * 4);
    float*    agg    = (float*)   alloc((size_t)N * HID * 4);
    float*    alphaB = (float*)   alloc((size_t)Etot * HEADS * 4);
    float*    als    = (float*)   alloc((size_t)N * HEADS * 4);
    float*    ald    = (float*)   alloc((size_t)N * HEADS * 4);
    float*    mBuf   = (float*)   alloc((size_t)N * HEADS * 4);
    float*    den    = (float*)   alloc((size_t)N * HEADS * 4);
    float*    sumA   = (float*)   alloc((size_t)N * EDGE_DIM * 4);
    float*    meanA  = (float*)   alloc((size_t)N * EDGE_DIM * 4);
    float*    cnt    = (float*)   alloc((size_t)N * 4);
    _Float16* WtG    = (_Float16*)alloc(64 * 64 * 2);
    float*    weff   = (float*)   alloc(EDGE_DIM * HEADS * 4);
    float*    pooled = (float*)   alloc(NUM_GRAPHS * HID * 4);
    float*    gcnt   = (float*)   alloc(NUM_GRAPHS * 4);

    const int B = 256;
    auto gb = [](long long n, int b) { return (unsigned)((n + b - 1) / b); };

    // ---- self-loop mean edge attr
    hipMemsetAsync(cnt,  0, (size_t)N * 4, stream);
    hipMemsetAsync(sumA, 0, (size_t)N * EDGE_DIM * 4, stream);
    hipMemsetAsync(pooled, 0, NUM_GRAPHS * HID * 4, stream);
    hipMemsetAsync(gcnt,   0, NUM_GRAPHS * 4, stream);
    k_edge_degree<<<gb(E, B), B, 0, stream>>>(dst, ea, cnt, sumA, E);
    k_mean_attr<<<gb((long long)N * EDGE_DIM, B), B, 0, stream>>>(sumA, cnt, meanA, N);

    // ---- layer-0 input in f16
    k_f32_to_f16<<<gb((long long)N * NUM_FEAT, B), B, 0, stream>>>(x, xh16, N * NUM_FEAT);

    for (int L = 0; L < 3; ++L) {
        const float* W  = (L == 0) ? W0  : Wh   + (size_t)(L - 1) * HID * HID;
        const float* as = (L == 0) ? as0 : as_h + (size_t)(L - 1) * HEADS * GAT_DIM;
        const float* ad = (L == 0) ? ad0 : ad_h + (size_t)(L - 1) * HEADS * GAT_DIM;
        const float* We = (L == 0) ? We0 : We_h + (size_t)(L - 1) * EDGE_DIM * HID;
        const float* ae = (L == 0) ? ae0 : ae_h + (size_t)(L - 1) * HEADS * GAT_DIM;
        const float* bb = (L == 0) ? b0  : bh   + (size_t)(L - 1) * HID;
        const _Float16* Ain = (L == 0) ? xh16 : hF16;
        int Kdim = (L == 0) ? NUM_FEAT : HID;

        k_prep_wt<<<gb(64 * Kdim, B), B, 0, stream>>>(W, WtG, Kdim);
        k_prep_weff<<<1, 64, 0, stream>>>(We, ae, weff);

        // xp = Ain @ W   (WMMA, 8 waves/block, 16 rows per wave)
        k_gemm_wmma<<<gb((long long)(N / 16), 8), 256, 0, stream>>>(Ain, WtG, xp, N, Kdim);

        k_node_logits<<<gb(N, B), B, 0, stream>>>(xp, as, ad, als, ald, N);
        k_layer_init<<<gb((long long)N * HID, B), B, 0, stream>>>(mBuf, den, agg, N);

        k_edge_alpha<<<gb(Etot, B), B, 0, stream>>>(src, dst, ea, meanA, als, ald, weff,
                                                    alphaB, mBuf, E, Etot);
        k_edge_exp<<<gb(Etot, B), B, 0, stream>>>(dst, alphaB, mBuf, den, E, Etot);
        k_edge_aggregate<<<gb((long long)Etot * HID, B), B, 0, stream>>>(
            src, dst, alphaB, den, xp, agg, E, Etot);

        k_finalize<<<gb((long long)N * HID, B), B, 0, stream>>>(agg, bb, hF32, hF16, N,
                                                                (L == 0) ? 1 : 0);
    }

    // ---- global mean pool + MLP head
    k_pool<<<gb((long long)N * HID, B), B, 0, stream>>>(hF32, batch, pooled, gcnt, N);
    k_mlp_head<<<1, 64, 0, stream>>>(pooled, gcnt, fc1w, fc1b, fc2w, fc2b, y_out);
}
